// Contrast_Item_13280038879821
// MI455X (gfx1250) — matrix operand order repaired and checked by reading, not compiled, not run on hardware
//
#include <hip/hip_runtime.h>
#include <hip/hip_bf16.h>
#include <math.h>

// ---------------------------------------------------------------------------
// Problem constants (match reference)
// ---------------------------------------------------------------------------
#define NTOT     8192
#define IN_SIZE  256
#define HID      128
#define INV_TEMP 5.0f          // 1 / 0.2
#define EPS      1e-8f

typedef __attribute__((ext_vector_type(16))) _Float16 v16h;
typedef __attribute__((ext_vector_type(8)))  _Float16 h8;
typedef __attribute__((ext_vector_type(8)))  float    v8f;

// ---------------------------------------------------------------------------
// Kernel 1: zp = elu(z @ W1 + b1) @ W2 + b2 ; row-L2-normalize ; cast to f16.
// One block (128 threads) per row.
// ---------------------------------------------------------------------------
__global__ void proj_norm_kernel(const float* __restrict__ z,
                                 const float* __restrict__ W1,
                                 const float* __restrict__ b1,
                                 const float* __restrict__ W2,
                                 const float* __restrict__ b2,
                                 _Float16* __restrict__ out)
{
    __shared__ float zrow[IN_SIZE];
    __shared__ float h1s[HID];
    __shared__ float red[HID];

    const int row = blockIdx.x;
    const int t   = threadIdx.x;           // 0..127

    const float* zr = z + (size_t)row * IN_SIZE;
    zrow[t]       = zr[t];
    zrow[t + HID] = zr[t + HID];
    __syncthreads();

    // Linear 1 + ELU
    float acc = b1[t];
#pragma unroll 4
    for (int k = 0; k < IN_SIZE; ++k)
        acc = fmaf(zrow[k], W1[k * HID + t], acc);
    float h = (acc > 0.0f) ? acc : expm1f(acc);
    h1s[t] = h;
    __syncthreads();

    // Linear 2
    float acc2 = b2[t];
#pragma unroll 4
    for (int k = 0; k < HID; ++k)
        acc2 = fmaf(h1s[k], W2[k * HID + t], acc2);

    // L2 norm over the 128 outputs
    red[t] = acc2 * acc2;
    __syncthreads();
    for (int s = HID / 2; s > 0; s >>= 1) {
        if (t < s) red[t] += red[t + s];
        __syncthreads();
    }
    const float nrm   = sqrtf(red[0]);
    const float scale = 1.0f / fmaxf(nrm, 1e-12f);

    out[(size_t)row * HID + t] = (_Float16)(acc2 * scale);
}

// ---------------------------------------------------------------------------
// Kernel 2: zero the accumulator arrays.
// ---------------------------------------------------------------------------
__global__ void zero_kernel(float* __restrict__ p, int n)
{
    int i = blockIdx.x * blockDim.x + threadIdx.x;
    if (i < n) p[i] = 0.0f;
}

// ---------------------------------------------------------------------------
// Kernel 3: WMMA row reduction over sim = exp( (A @ B^T) * INV_TEMP ).
//   rowsum[i] += sum_j sim[i,j]
//   rowpos[i] += sum_j sim[i,j] * pos[i*NTOT + j]
// A, B: [NTOT x 128] f16 (already L2-normalized). One wave = 16-row strip.
// Block = 256 threads (8 waves) = 128 rows; grid.y splits the column range.
// ---------------------------------------------------------------------------
__global__ void sim_row_kernel(const _Float16* __restrict__ A,
                               const _Float16* __restrict__ B,
                               const int*      __restrict__ pos,
                               float* __restrict__ rowsum,
                               float* __restrict__ rowpos,
                               int jt_per_block)
{
    const int lane    = threadIdx.x & 31;
    const int wave    = threadIdx.x >> 5;
    const int rowBase = blockIdx.x * 128 + wave * 16;
    const int m       = lane & 15;          // A: row-in-tile / B: col-in-tile
    const int g       = lane >> 4;          // half-wave group
    const int jt0     = blockIdx.y * jt_per_block;

    // --- Load A fragments once: 16x32 f16 layout per ISA (4 K-steps) -------
    const _Float16* arow = A + (size_t)(rowBase + m) * HID;
    v16h afrag[4];
#pragma unroll
    for (int s = 0; s < 4; ++s) {
        const int kb = s * 32;
        h8 lo = *(const h8*)(arow + kb + g * 8);
        h8 hi = *(const h8*)(arow + kb + 16 + g * 8);
#pragma unroll
        for (int i = 0; i < 8; ++i) { afrag[s][i] = lo[i]; afrag[s][8 + i] = hi[i]; }
    }

    float racc[8], rpacc[8];
#pragma unroll
    for (int r = 0; r < 8; ++r) { racc[r] = 0.0f; rpacc[r] = 0.0f; }

    for (int jt = 0; jt < jt_per_block; ++jt) {
        const int jbase = (jt0 + jt) * 16;
        // B fragment: column n = B-row (jbase + n); two 8-half K runs (table order).
        const _Float16* brow = B + (size_t)(jbase + m) * HID;

        v8f c = {0.f, 0.f, 0.f, 0.f, 0.f, 0.f, 0.f, 0.f};
#pragma unroll
        for (int s = 0; s < 4; ++s) {
            const int kb = s * 32 + g * 8;
            h8 lo = *(const h8*)(brow + kb);
            h8 hi = *(const h8*)(brow + kb + 16);
            v16h bfrag;
#pragma unroll
            for (int i = 0; i < 8; ++i) { bfrag[i] = lo[i]; bfrag[8 + i] = hi[i]; }
            c = __builtin_amdgcn_wmma_f32_16x16x32_f16(
                    false, afrag[s], false, bfrag, (short)0, c, false, false);
        }

        // c[r] holds dot for M = rowBase + r + 8*g, N = jbase + m
        const int ncol = jbase + m;
        const int* pcol = pos + (size_t)(rowBase + 8 * g) * NTOT + ncol;
#pragma unroll
        for (int r = 0; r < 8; ++r) {
            const float s  = __expf(c[r] * INV_TEMP);
            const float pf = (float)pcol[(size_t)r * NTOT];
            racc[r]  += s;
            rpacc[r]  = fmaf(s, pf, rpacc[r]);
        }
    }

    // Reduce across the 16 lanes sharing each M (xor masks 1,2,4,8),
    // then one atomic per row per block.
#pragma unroll
    for (int r = 0; r < 8; ++r) {
        float a = racc[r];
        float b = rpacc[r];
        for (int mask = 1; mask < 16; mask <<= 1) {
            a += __shfl_xor(a, mask, 32);
            b += __shfl_xor(b, mask, 32);
        }
        if (m == 0) {
            atomicAdd(rowsum + rowBase + r + 8 * g, a);
            atomicAdd(rowpos + rowBase + r + 8 * g, b);
        }
    }
}

// ---------------------------------------------------------------------------
// Kernel 4: final per-row loss.
//   lori_mp = -log(rowpos1/(rowsum1+EPS) + EPS)
//   lori_sc = -log(rowpos2/(rowsum2+EPS) + EPS)
//   ans = 0.5*lori_mp + 0.5*lori_sc ; NaN/Inf -> 0
// ---------------------------------------------------------------------------
__global__ void loss_kernel(const float* __restrict__ rowsum1,
                            const float* __restrict__ rowpos1,
                            const float* __restrict__ rowsum2,
                            const float* __restrict__ rowpos2,
                            float* __restrict__ out, int n)
{
    int i = blockIdx.x * blockDim.x + threadIdx.x;
    if (i >= n) return;
    const float a1 = rowpos1[i] / (rowsum1[i] + EPS) + EPS;
    const float a2 = rowpos2[i] / (rowsum2[i] + EPS) + EPS;
    float v = 0.5f * (-logf(a1)) + 0.5f * (-logf(a2));
    if (!isfinite(v)) v = 0.0f;
    out[i] = v;
}

// ---------------------------------------------------------------------------
// Launch
// ---------------------------------------------------------------------------
extern "C" void kernel_launch(void* const* d_in, const int* in_sizes, int n_in,
                              void* d_out, int out_size, void* d_ws, size_t ws_size,
                              hipStream_t stream)
{
    const float* z_mp = (const float*)d_in[0];
    const float* z_sc = (const float*)d_in[1];
    const float* W1   = (const float*)d_in[2];
    const float* b1   = (const float*)d_in[3];
    const float* W2   = (const float*)d_in[4];
    const float* b2   = (const float*)d_in[5];
    const int*   pos  = (const int*)d_in[6];
    float*       out  = (float*)d_out;

    // Workspace layout
    char* ws = (char*)d_ws;
    _Float16* Amp = (_Float16*)(ws);                                   // 2 MB
    _Float16* Bsc = (_Float16*)(ws + (size_t)NTOT * HID * 2);          // 2 MB
    float* rowsum1 = (float*)(ws + (size_t)2 * NTOT * HID * 2);
    float* rowpos1 = rowsum1 + NTOT;
    float* rowsum2 = rowpos1 + NTOT;
    float* rowpos2 = rowsum2 + NTOT;

    // 1) Projections + L2 normalize + f16 cast
    proj_norm_kernel<<<NTOT, HID, 0, stream>>>(z_mp, W1, b1, W2, b2, Amp);
    proj_norm_kernel<<<NTOT, HID, 0, stream>>>(z_sc, W1, b1, W2, b2, Bsc);

    // 2) Zero accumulators (ws is poisoned by the harness)
    zero_kernel<<<(4 * NTOT + 255) / 256, 256, 0, stream>>>(rowsum1, 4 * NTOT);

    // 3) WMMA passes. Grid: 64 row-blocks x 8 column splits; 512 tiles / 8 = 64.
    dim3 grid(NTOT / 128, 8, 1);
    const int jt_per_block = (NTOT / 16) / 8;
    //   pass 1: rows = mp  -> rowsum(sim), rowpos(sim)   [lori_mp side]
    sim_row_kernel<<<grid, 256, 0, stream>>>(Amp, Bsc, pos, rowsum1, rowpos1, jt_per_block);
    //   pass 2: rows = sc  -> colsum(sim), colpos(sim)   [lori_sc side]
    sim_row_kernel<<<grid, 256, 0, stream>>>(Bsc, Amp, pos, rowsum2, rowpos2, jt_per_block);

    // 4) Final loss
    loss_kernel<<<(NTOT + 255) / 256, 256, 0, stream>>>(rowsum1, rowpos1,
                                                        rowsum2, rowpos2, out, NTOT);
}
